// Seq2Seq_5265629905441
// MI455X (gfx1250) — matrix operand order repaired and checked by reading, not compiled, not run on hardware
//
#include <hip/hip_runtime.h>
#include <stdint.h>

// ---------------------------------------------------------------------------
// Seq2Seq LSTM for MI455X (gfx1250, wave32, WMMA).
//
// Roofline reasoning: 608 sequential LSTM steps, each ~0.5-1 GFLOP of GEMM on
// a 64-row activation matrix. Weights (~52 MB packed bf16) are L2-resident
// (192 MB L2), so each step is latency-bound. We therefore:
//   * pre-pack all weights once into the native WMMA B-fragment layout so a
//     B fragment is two global_load_b128 per lane,
//   * run gates = x@Wih^T + h@Whh^T + bias fused in one WMMA kernel per step
//     (f32 accumulation, bf16 operands -> v_wmma_f32_16x16x32_bf16),
//   * stage the shared A (activation) fragments through LDS in WMMA lane
//     layout (cooperative fill, ds_load_b128 consume),
//   * small elementwise kernel for the cell nonlinearity (f32 cell state).
// ---------------------------------------------------------------------------

typedef __attribute__((ext_vector_type(16))) __bf16 v16bf;
typedef __attribute__((ext_vector_type(8)))  float  v8f;
typedef __attribute__((ext_vector_type(4)))  int    v4i;

typedef unsigned short bf16s;   // raw bf16 storage

#define BSZ  64
#define SEQ  256
#define IDIM 256
#define HDIM 1024
#define G4H  4096
#define TDEC 96

union Frag { v4i i[2]; v16bf v; };

__device__ __forceinline__ bf16s f2bf(float f) {
  union { float f; unsigned u; } c; c.f = f;
  unsigned u = c.u;
  return (bf16s)((u + 0x7FFFu + ((u >> 16) & 1u)) >> 16);   // RNE
}

__device__ __forceinline__ float sigmoidf_(float x) {
  return 1.0f / (1.0f + __expf(-x));
}

// --- one-time prep kernels --------------------------------------------------

// Pack f32 weight W[N][K] (row-major, N = output features) into bf16 WMMA
// B-fragment layout: tile (kt, nt) of 32x16, lane l holds n = nt*16 + (l&15),
// k = kt*32 + (l>=16 ? 8 : 0) + (e&7) + (e>=8 ? 16 : 0) for e = 0..15.
__global__ __launch_bounds__(256) void pack_kernel(
    const float* __restrict__ W, bf16s* __restrict__ Wp, int N, int K) {
  const int tid = blockIdx.x * 256 + threadIdx.x;
  if (tid >= N * K) return;
  const int e       = tid & 15;
  const int lane    = (tid >> 4) & 31;
  const int tileIdx = tid >> 9;
  const int ntiles  = N >> 4;
  const int nt = tileIdx % ntiles;
  const int kt = tileIdx / ntiles;
  const int n  = nt * 16 + (lane & 15);
  const int k  = kt * 32 + ((lane >> 4) << 3) + (e & 7) + ((e >> 3) << 4);
  Wp[tid] = f2bf(W[(size_t)n * K + k]);
}

__global__ __launch_bounds__(256) void bias_kernel(
    const float* __restrict__ a, const float* __restrict__ b,
    float* __restrict__ o) {
  const int i = blockIdx.x * 256 + threadIdx.x;   // < 4096
  o[i] = a[i] + b[i];
}

// input_seq (B,S,I) f32 -> time-major bf16 (S,B,I)
__global__ __launch_bounds__(256) void conv_in_kernel(
    const float* __restrict__ in, bf16s* __restrict__ o) {
  const int tid = blockIdx.x * 256 + threadIdx.x;   // < SEQ*BSZ*IDIM
  const int s = tid >> 14;                           // BSZ*IDIM = 16384
  const int r = tid & 16383;
  const int b = r >> 8;
  const int i = r & 255;
  o[tid] = f2bf(in[((size_t)b * SEQ + s) * IDIM + i]);
}

__global__ __launch_bounds__(256) void init_state_kernel(
    const float* __restrict__ h0, const float* __restrict__ c0,
    bf16s* __restrict__ hA, float* __restrict__ cA,
    bf16s* __restrict__ hB, float* __restrict__ cB) {
  const int idx = blockIdx.x * 256 + threadIdx.x;   // < 65536 = B*H
  hA[idx] = f2bf(h0[idx]);
  cA[idx] = c0[idx];
  hB[idx] = f2bf(h0[BSZ * HDIM + idx]);
  cB[idx] = c0[BSZ * HDIM + idx];
}

// --- per-step fused gate GEMM ----------------------------------------------
// gates[64][4096] = x[64][Kx] @ Wx^T + h[64][1024] @ Wh^T + bias
// Block: 256 threads = 8 waves; wave w -> mt = w&3 (4 M-tiles cover B=64),
// nt = blockIdx.x*2 + (w>>2). Grid.x = 128 covers all 256 N-tiles.
// A fragments (shared by every nt) staged via LDS in WMMA lane layout.
__global__ __launch_bounds__(256) void gates_kernel(
    const bf16s* __restrict__ x, int Kx, const bf16s* __restrict__ Wx,
    const bf16s* __restrict__ h, const bf16s* __restrict__ Wh,
    const float* __restrict__ bias, float* __restrict__ gates) {
  __shared__ v4i ldsA[4][4][32][2];   // [mt][kt-in-chunk][lane][2x16B] = 16 KB
  const int tid  = threadIdx.x;
  const int lane = tid & 31;
  const int w    = tid >> 5;
  const int mt   = w & 3;
  const int nt   = blockIdx.x * 2 + (w >> 2);

  v8f acc = {};

  for (int phase = 0; phase < 2; ++phase) {
    const bf16s* A = phase ? h : x;
    const int    K = phase ? HDIM : Kx;
    const bf16s* W = phase ? Wh : Wx;
    const int ktiles = K >> 5;
    for (int kc = 0; kc < ktiles; kc += 4) {
      __syncthreads();
      // Cooperative fill: 4 mt * 4 kt * 32 lanes = 512 fragments-rows,
      // 2 per thread. A-fragment lane layout (16-bit A 16x32):
      //   m = lane&15, base = (lane<16 ? 0 : 8),
      //   halves 0..7  -> K = base+0..7   (contiguous 16B)
      //   halves 8..15 -> K = base+16..23 (contiguous 16B)
      #pragma unroll
      for (int r = 0; r < 2; ++r) {
        const int ent = r * 256 + tid;       // 0..511
        const int el  = ent & 31;
        const int ekt = (ent >> 5) & 3;
        const int emt = ent >> 7;
        const int m     = emt * 16 + (el & 15);
        const int kbase = (kc + ekt) * 32 + ((el >> 4) << 3);  // +0 or +8
        const bf16s* src = A + (size_t)m * K + kbase;
        ldsA[emt][ekt][el][0] = *(const v4i*)(src);
        ldsA[emt][ekt][el][1] = *(const v4i*)(src + 16);
      }
      __syncthreads();
      #pragma unroll
      for (int kk = 0; kk < 4; ++kk) {
        const int kt = kc + kk;
        Frag fa, fb;
        fa.i[0] = ldsA[mt][kk][lane][0];
        fa.i[1] = ldsA[mt][kk][lane][1];
        const v4i* wp = (const v4i*)W +
            ((size_t)(kt * (G4H >> 4) + nt) * 32 + lane) * 2;
        fb.i[0] = wp[0];
        fb.i[1] = wp[1];
        acc = __builtin_amdgcn_wmma_f32_16x16x32_bf16(
            false, fa.v, false, fb.v, (short)0, acc, false, false);
      }
    }
  }

  // C/D layout: lane l, elem j: n = l&15, m = (l<16 ? j : 8+j)
  const int   n  = nt * 16 + (lane & 15);
  const float bn = bias[n];
  const int   m0 = mt * 16 + ((lane < 16) ? 0 : 8);
  #pragma unroll
  for (int j = 0; j < 8; ++j)
    gates[(size_t)(m0 + j) * G4H + n] = acc[j] + bn;
}

// --- elementwise LSTM cell update -------------------------------------------
__global__ __launch_bounds__(256) void cell_kernel(
    const float* __restrict__ gates, float* __restrict__ c,
    bf16s* __restrict__ hst, bf16s* __restrict__ ys) {
  const int idx = blockIdx.x * 256 + threadIdx.x;   // < 65536
  const int b = idx >> 10;
  const int j = idx & 1023;
  const float* g = gates + (size_t)b * G4H;
  const float gi = g[j];
  const float gf = g[HDIM + j];
  const float gg = g[2 * HDIM + j];
  const float go = g[3 * HDIM + j];
  const float cn = sigmoidf_(gf) * c[idx] + sigmoidf_(gi) * tanhf(gg);
  const float hn = sigmoidf_(go) * tanhf(cn);
  c[idx] = cn;
  const bf16s hb = f2bf(hn);
  hst[idx] = hb;
  if (ys) ys[idx] = hb;
}

// --- decoder output projection: pred = hB @ lin_W^T + lin_b -----------------
// 64x256x1024 GEMM; 64 waves (8 blocks). Writes bf16 feedback and column 0
// of the prediction directly into d_out[b*T + t].
__global__ __launch_bounds__(256) void linear_kernel(
    const bf16s* __restrict__ hB, const bf16s* __restrict__ Wp,
    const float* __restrict__ lb, bf16s* __restrict__ pred,
    float* __restrict__ out, int t) {
  const int tid  = threadIdx.x;
  const int lane = tid & 31;
  const int w    = tid >> 5;
  const int tile = blockIdx.x * 8 + w;   // 0..63
  const int nt = tile & 15;
  const int mt = tile >> 4;
  const int m  = mt * 16 + (lane & 15);
  const int kb = (lane < 16) ? 0 : 8;
  v8f acc = {};
  for (int kt = 0; kt < (HDIM >> 5); ++kt) {
    Frag fa, fb;
    const bf16s* src = hB + (size_t)m * HDIM + kt * 32 + kb;
    fa.i[0] = *(const v4i*)src;
    fa.i[1] = *(const v4i*)(src + 16);
    const v4i* wp = (const v4i*)Wp + ((size_t)(kt * 16 + nt) * 32 + lane) * 2;
    fb.i[0] = wp[0];
    fb.i[1] = wp[1];
    acc = __builtin_amdgcn_wmma_f32_16x16x32_bf16(
        false, fa.v, false, fb.v, (short)0, acc, false, false);
  }
  const int   n  = nt * 16 + (lane & 15);
  const float bn = lb[n];
  const int   m0 = mt * 16 + ((lane < 16) ? 0 : 8);
  #pragma unroll
  for (int j = 0; j < 8; ++j) {
    const float v = acc[j] + bn;
    pred[(size_t)(m0 + j) * IDIM + n] = f2bf(v);
    if (n == 0) out[(m0 + j) * TDEC + t] = v;
  }
}

// ---------------------------------------------------------------------------

extern "C" void kernel_launch(void* const* d_in, const int* in_sizes, int n_in,
                              void* d_out, int out_size, void* d_ws, size_t ws_size,
                              hipStream_t stream) {
  (void)in_sizes; (void)n_in; (void)out_size; (void)ws_size;

  const float* input_seq = (const float*)d_in[0];
  const float* h0        = (const float*)d_in[1];
  const float* c0        = (const float*)d_in[2];
  const float* encWih0   = (const float*)d_in[3];
  const float* encWhh0   = (const float*)d_in[4];
  const float* encBih0   = (const float*)d_in[5];
  const float* encBhh0   = (const float*)d_in[6];
  const float* encWih1   = (const float*)d_in[7];
  const float* encWhh1   = (const float*)d_in[8];
  const float* encBih1   = (const float*)d_in[9];
  const float* encBhh1   = (const float*)d_in[10];
  const float* decWih0   = (const float*)d_in[11];
  const float* decWhh0   = (const float*)d_in[12];
  const float* decBih0   = (const float*)d_in[13];
  const float* decBhh0   = (const float*)d_in[14];
  const float* decWih1   = (const float*)d_in[15];
  const float* decWhh1   = (const float*)d_in[16];
  const float* decBih1   = (const float*)d_in[17];
  const float* decBhh1   = (const float*)d_in[18];
  const float* linW      = (const float*)d_in[19];
  const float* linB      = (const float*)d_in[20];
  float* out = (float*)d_out;

  // workspace carve-out (256B aligned)
  char* ws = (char*)d_ws;
  size_t off = 0;
  auto carve = [&](size_t bytes) -> void* {
    void* p = ws + off;
    off = (off + bytes + 255) & ~(size_t)255;
    return p;
  };
  bf16s* pEncWih0 = (bf16s*)carve((size_t)G4H * IDIM * 2);
  bf16s* pEncWhh0 = (bf16s*)carve((size_t)G4H * HDIM * 2);
  bf16s* pEncWih1 = (bf16s*)carve((size_t)G4H * HDIM * 2);
  bf16s* pEncWhh1 = (bf16s*)carve((size_t)G4H * HDIM * 2);
  bf16s* pDecWih0 = (bf16s*)carve((size_t)G4H * IDIM * 2);
  bf16s* pDecWhh0 = (bf16s*)carve((size_t)G4H * HDIM * 2);
  bf16s* pDecWih1 = (bf16s*)carve((size_t)G4H * HDIM * 2);
  bf16s* pDecWhh1 = (bf16s*)carve((size_t)G4H * HDIM * 2);
  bf16s* pLin     = (bf16s*)carve((size_t)IDIM * HDIM * 2);
  float* bEnc0 = (float*)carve(G4H * 4);
  float* bEnc1 = (float*)carve(G4H * 4);
  float* bDec0 = (float*)carve(G4H * 4);
  float* bDec1 = (float*)carve(G4H * 4);
  bf16s* inbf  = (bf16s*)carve((size_t)SEQ * BSZ * IDIM * 2);
  bf16s* ys    = (bf16s*)carve((size_t)SEQ * BSZ * HDIM * 2);
  bf16s* hA    = (bf16s*)carve((size_t)BSZ * HDIM * 2);
  float* cA    = (float*)carve((size_t)BSZ * HDIM * 4);
  bf16s* hB    = (bf16s*)carve((size_t)BSZ * HDIM * 2);
  float* cB    = (float*)carve((size_t)BSZ * HDIM * 4);
  float* gates = (float*)carve((size_t)BSZ * G4H * 4);
  bf16s* pred  = (bf16s*)carve((size_t)BSZ * IDIM * 2);

  const dim3 blk(256);

  // ---- one-time prep ----
  auto pack = [&](const float* W, bf16s* Wp, int N, int K) {
    pack_kernel<<<dim3((N * K) / 256), blk, 0, stream>>>(W, Wp, N, K);
  };
  pack(encWih0, pEncWih0, G4H, IDIM);
  pack(encWhh0, pEncWhh0, G4H, HDIM);
  pack(encWih1, pEncWih1, G4H, HDIM);
  pack(encWhh1, pEncWhh1, G4H, HDIM);
  pack(decWih0, pDecWih0, G4H, IDIM);
  pack(decWhh0, pDecWhh0, G4H, HDIM);
  pack(decWih1, pDecWih1, G4H, HDIM);
  pack(decWhh1, pDecWhh1, G4H, HDIM);
  pack(linW,    pLin,     IDIM, HDIM);

  bias_kernel<<<dim3(G4H / 256), blk, 0, stream>>>(encBih0, encBhh0, bEnc0);
  bias_kernel<<<dim3(G4H / 256), blk, 0, stream>>>(encBih1, encBhh1, bEnc1);
  bias_kernel<<<dim3(G4H / 256), blk, 0, stream>>>(decBih0, decBhh0, bDec0);
  bias_kernel<<<dim3(G4H / 256), blk, 0, stream>>>(decBih1, decBhh1, bDec1);

  conv_in_kernel<<<dim3((SEQ * BSZ * IDIM) / 256), blk, 0, stream>>>(input_seq, inbf);
  init_state_kernel<<<dim3((BSZ * HDIM) / 256), blk, 0, stream>>>(h0, c0, hA, cA, hB, cB);

  const dim3 gGrid(128);   // 256 N-tiles / 2 per block
  const dim3 cGrid((BSZ * HDIM) / 256);

  // ---- encoder layer 0 (stores ys for layer 1) ----
  for (int s = 0; s < SEQ; ++s) {
    gates_kernel<<<gGrid, blk, 0, stream>>>(
        inbf + (size_t)s * BSZ * IDIM, IDIM, pEncWih0, hA, pEncWhh0, bEnc0, gates);
    cell_kernel<<<cGrid, blk, 0, stream>>>(gates, cA, hA, ys + (size_t)s * BSZ * HDIM);
  }
  // ---- encoder layer 1 ----
  for (int s = 0; s < SEQ; ++s) {
    gates_kernel<<<gGrid, blk, 0, stream>>>(
        ys + (size_t)s * BSZ * HDIM, HDIM, pEncWih1, hB, pEncWhh1, bEnc1, gates);
    cell_kernel<<<cGrid, blk, 0, stream>>>(gates, cB, hB, (bf16s*)nullptr);
  }
  // ---- decoder (states hA/cA, hB/cB carry over from encoder finals) ----
  const bf16s* dec_x = inbf + (size_t)(SEQ - 1) * BSZ * IDIM;   // input_seq[:, -1, :]
  for (int t = 0; t < TDEC; ++t) {
    gates_kernel<<<gGrid, blk, 0, stream>>>(
        dec_x, IDIM, pDecWih0, hA, pDecWhh0, bDec0, gates);
    cell_kernel<<<cGrid, blk, 0, stream>>>(gates, cA, hA, (bf16s*)nullptr);
    gates_kernel<<<gGrid, blk, 0, stream>>>(
        hA, HDIM, pDecWih1, hB, pDecWhh1, bDec1, gates);
    cell_kernel<<<cGrid, blk, 0, stream>>>(gates, cB, hB, (bf16s*)nullptr);
    linear_kernel<<<dim3(8), blk, 0, stream>>>(hB, pLin, linB, pred, out, t);
    dec_x = pred;   // feedback
  }
}
